// ComplexVariableWidthCNNRNN_77429670412525
// MI455X (gfx1250) — matrix operand order repaired and checked by reading, not compiled, not run on hardware
//
#include <hip/hip_runtime.h>
#include <math.h>

// ---------------- problem constants ----------------
constexpr int Bc = 16, Tc = 96, Hc = 31, Wc = 64;
constexpr int C1c = 16, C2c = 32, LHc = 64, NCc = 10;
constexpr int BT = Bc * Tc;          // 1536
constexpr int PIX = Hc * Wc;         // 1984 = 124 * 16
constexpr int K2 = C1c * 9;          // 144
constexpr int K2P = 160;             // padded to 5 chunks of 32
constexpr int NG = 4 * LHc;          // 256 gates

typedef _Float16 v16h __attribute__((ext_vector_type(16)));
typedef float    v8f  __attribute__((ext_vector_type(8)));
typedef float    v2f  __attribute__((ext_vector_type(2)));
typedef unsigned int u32x4 __attribute__((ext_vector_type(4)));
typedef int          i32x8 __attribute__((ext_vector_type(8)));
typedef int          i32x4 __attribute__((ext_vector_type(4)));

__device__ __forceinline__ float sigmoidf_(float x) { return 1.0f / (1.0f + expf(-x)); }
__device__ __forceinline__ int clampi_(int v, int lo, int hi) {
  return v < lo ? lo : (v > hi ? hi : v);
}

// ============================================================
// Kernel 1: per-image conv1(1->16,relu) -> conv2(16->32,relu, WMMA f16)
//           -> mean pool, all in LDS. One block per image.
//           Input plane staged into LDS by the Tensor Data Mover.
// ============================================================
__global__ __launch_bounds__(256) void conv_pool_kernel(
    const float* __restrict__ x,   // [BT, 31, 64]
    const float* __restrict__ w1,  // [16, 1, 3, 3]
    const float* __restrict__ b1,  // [16]
    const float* __restrict__ w2,  // [32, 16, 3, 3] == [32][144]
    const float* __restrict__ b2,  // [32]
    float* __restrict__ pooled)    // [BT, 32]
{
  __shared__ float     sIn[PIX];             // input plane f32 (TDM target)
  __shared__ _Float16  sAct[C1c * PIX];      // conv1 output (relu), f16
  __shared__ _Float16  sW2[C2c * K2P];       // conv2 weights, f16, K padded
  __shared__ float     sW1[C1c * 9];
  __shared__ float     sB1[C1c];
  __shared__ float     sB2[C2c];
  __shared__ float     sPool[C2c];
  __shared__ int       sKBase[K2P];
  __shared__ signed char sKDy[K2P], sKDx[K2P];

  const int tid = threadIdx.x;
  const int img = blockIdx.x;

  // ---- TDM: DMA the 31x64 f32 input plane into LDS (1D tile) ----
  if (tid == 0) {
    unsigned int lds_off = (unsigned int)(size_t)(void*)&sIn[0];
    unsigned long long ga = (unsigned long long)(const void*)(x + (size_t)img * PIX);
    u32x4 g0;
    g0[0] = 1u;                                        // count=1 (user mode)
    g0[1] = lds_off;                                   // lds_addr
    g0[2] = (unsigned int)(ga & 0xFFFFFFFFu);          // global_addr[31:0]
    g0[3] = (unsigned int)((ga >> 32) & 0x01FFFFFFu)   // global_addr[56:32]
            | (2u << 30);                              // type=2 (image)
    i32x8 g1;
    g1[0] = 0x00020000;          // wg_mask=0, data_size=2 (4 bytes)
    g1[1] = (int)(PIX << 16);    // tensor_dim0[15:0] in bits[63:48]
    g1[2] = 0x00010000;          // tensor_dim0[31:16]=0, tensor_dim1=1
    g1[3] = (int)(PIX << 16);    // tensor_dim1 hi=0, tile_dim0=PIX
    g1[4] = 1;                   // tile_dim1=1, tile_dim2=0
    g1[5] = PIX;                 // tensor_dim0_stride[31:0]
    g1[6] = 0;                   // stride hi, tensor_dim1_stride lo16
    g1[7] = 0;
    i32x4 z4 = {};
    i32x8 z8 = {};
    __builtin_amdgcn_tensor_load_to_lds(g0, g1, z4, z4, z8, 0);
  }
  __builtin_amdgcn_s_wait_tensorcnt(0);

  // ---- stage weights / tables ----
  for (int i = tid; i < C1c * 9; i += 256) sW1[i] = w1[i];
  if (tid < C1c) sB1[tid] = b1[tid];
  if (tid < C2c) { sB2[tid] = b2[tid]; sPool[tid] = 0.0f; }
  for (int i = tid; i < C2c * K2P; i += 256) {
    int co = i / K2P, k = i % K2P;
    sW2[i] = (k < K2) ? (_Float16)w2[co * K2 + k] : (_Float16)0.0f;
  }
  for (int k = tid; k < K2P; k += 256) {
    if (k < K2) {
      int ci = k / 9, r = k % 9;
      sKBase[k] = ci * PIX;
      sKDy[k] = (signed char)(r / 3 - 1);
      sKDx[k] = (signed char)(r % 3 - 1);
    } else { sKBase[k] = 0; sKDy[k] = 99; sKDx[k] = 99; }
  }
  __syncthreads();

  // ---- conv1 (cin=1) + bias + relu, store f16 (branchless halo) ----
  for (int idx = tid; idx < C1c * PIX; idx += 256) {
    int c = idx / PIX, p = idx % PIX;
    int py = p >> 6, px = p & 63;
    float acc = sB1[c];
#pragma unroll
    for (int t9 = 0; t9 < 9; ++t9) {
      int dy = t9 / 3 - 1, dx = t9 % 3 - 1;
      int yy = py + dy, xx = px + dx;
      bool ok = ((unsigned)yy < (unsigned)Hc) && ((unsigned)xx < (unsigned)Wc);
      float v = sIn[clampi_(yy, 0, Hc - 1) * Wc + clampi_(xx, 0, Wc - 1)];
      acc += ok ? sW1[c * 9 + t9] * v : 0.0f;
    }
    sAct[c * PIX + p] = (_Float16)fmaxf(acc, 0.0f);
  }
  __syncthreads();

  // ---- conv2 as implicit GEMM: [32 x 144] @ [144 x 16px] via WMMA f16 ----
  const int lane = tid & 31;
  const int wv   = __builtin_amdgcn_readfirstlane(tid >> 5);  // scalar wave id
  const int hi = (lane >> 4) & 1;
  const int nn = lane & 15;

  float psum0[8], psum1[8];
#pragma unroll
  for (int v = 0; v < 8; ++v) { psum0[v] = 0.0f; psum1[v] = 0.0f; }

  for (int tile = wv; tile < PIX / 16; tile += 8) {   // scalar loop bound
    const int p  = tile * 16 + nn;   // pixel handled by this lane (B-frag col)
    const int py = p >> 6, px = p & 63;
    v8f acc0 = {};
    v8f acc1 = {};
#pragma unroll
    for (int kc = 0; kc < K2P / 32; ++kc) {
      // B fragment (32x16, f16): lanes 0-15 hold K=0..7,16..23; hi lanes K=8..15,24..31.
      // Branchless: clamp address, select zero for halo / K-padding.
      v16h bf;
      const int klo = kc * 32 + hi * 8;
#pragma unroll
      for (int j = 0; j < 16; ++j) {
        int k = klo + (j < 8 ? j : j + 8);
        int yy = py + (int)sKDy[k];
        int xx = px + (int)sKDx[k];
        bool ok = ((unsigned)yy < (unsigned)Hc) && ((unsigned)xx < (unsigned)Wc)
                  && (k < K2);
        _Float16 v = sAct[sKBase[k] + clampi_(yy, 0, Hc - 1) * Wc
                                    + clampi_(xx, 0, Wc - 1)];
        bf[j] = ok ? v : (_Float16)0.0f;
      }
      // A fragments (16x32, f16): row m = lane&15; VGPR v holds K pairs
      v16h a0, a1;
#pragma unroll
      for (int v = 0; v < 8; ++v) {
        int kb = ((v < 4) ? 2 * v : 16 + 2 * (v - 4)) + hi * 8;
        int k  = kc * 32 + kb;
        a0[2 * v]     = sW2[nn * K2P + k];
        a0[2 * v + 1] = sW2[nn * K2P + k + 1];
        a1[2 * v]     = sW2[(16 + nn) * K2P + k];
        a1[2 * v + 1] = sW2[(16 + nn) * K2P + k + 1];
      }
      acc0 = __builtin_amdgcn_wmma_f32_16x16x32_f16(false, a0, false, bf, (short)0, acc0, false, false);
      acc1 = __builtin_amdgcn_wmma_f32_16x16x32_f16(false, a1, false, bf, (short)0, acc1, false, false);
    }
    // D layout: lane holds (m = v + hi*8, pixel = lane&15); bias+relu+pool
#pragma unroll
    for (int v = 0; v < 8; ++v) {
      int m = v + hi * 8;
      psum0[v] += fmaxf(acc0[v] + sB2[m], 0.0f);
      psum1[v] += fmaxf(acc1[v] + sB2[16 + m], 0.0f);
    }
  }
#pragma unroll
  for (int v = 0; v < 8; ++v) {
    int m = v + hi * 8;
    atomicAdd(&sPool[m],      psum0[v]);
    atomicAdd(&sPool[16 + m], psum1[v]);
  }
  __syncthreads();
  if (tid < C2c) pooled[(size_t)img * C2c + tid] = sPool[tid] * (1.0f / (float)PIX);
}

// ============================================================
// Kernel 2: feat = [r, m, sqrt(r^2+m^2), atan2(m,r)]  -> [BT,128]
// ============================================================
__global__ void feat_kernel(const float* __restrict__ pr,
                            const float* __restrict__ pm,
                            float* __restrict__ feat)
{
  int i = blockIdx.x * blockDim.x + threadIdx.x;
  if (i >= BT * C2c) return;
  int row = i / C2c, c = i % C2c;
  float r = pr[i], m = pm[i];
  float* f = feat + (size_t)row * (4 * C2c);
  f[c]            = r;
  f[C2c + c]      = m;
  f[2 * C2c + c]  = sqrtf(r * r + m * m);
  f[3 * C2c + c]  = atan2f(m, r);
}

// ============================================================
// Kernel 3: generic GEMM  C[M][N] = act(A[M][K] @ W[N][K]^T + bias[N])
//           pure-f32 WMMA 16x16x4 chain; one wave per 16x16 tile.
// ============================================================
__global__ __launch_bounds__(128) void gemm_wmma_f32(
    const float* __restrict__ A, const float* __restrict__ Wt,
    const float* __restrict__ bias, float* __restrict__ Cmat,
    int M, int N, int K, int relu)
{
  const int lane = threadIdx.x & 31;
  const int wv   = __builtin_amdgcn_readfirstlane(threadIdx.x >> 5);
  const int tile = blockIdx.x * 4 + wv;            // scalar
  const int ntN = N / 16;
  const int ntiles = (M / 16) * ntN;
  if (tile >= ntiles) return;                       // scalar branch
  const int m0 = (tile / ntN) * 16;
  const int n0 = (tile % ntN) * 16;
  const int hi = (lane >> 4) & 1;
  const int ln = lane & 15;

  v8f acc = {};
  for (int kc = 0; kc < K; kc += 4) {
    v2f a, b;
#pragma unroll
    for (int v = 0; v < 2; ++v) {
      int k = kc + hi * 2 + v;
      a[v] = A [(size_t)(m0 + ln) * K + k];   // A(m=ln, k)
      b[v] = Wt[(size_t)(n0 + ln) * K + k];   // B(k, n=ln) = W[n][k]
    }
    acc = __builtin_amdgcn_wmma_f32_16x16x4_f32(false, a, false, b, (short)0, acc, false, false);
  }
#pragma unroll
  for (int v = 0; v < 8; ++v) {
    int m = m0 + v + hi * 8;
    int n = n0 + ln;
    float val = acc[v] + bias[n];
    if (relu) val = fmaxf(val, 0.0f);
    Cmat[(size_t)m * N + n] = val;
  }
}

// ============================================================
// Kernel 4: build t-major LSTM inputs: xs_f[t][b] = fused[b][t],
//           xs_b[t][b] = fused[b][clip(len[b]-1-t,0,T-1)]
// ============================================================
__global__ void seq_prep_kernel(const float* __restrict__ fused,
                                const int* __restrict__ lengths,
                                float* __restrict__ xs_f,
                                float* __restrict__ xs_b)
{
  int i = blockIdx.x * blockDim.x + threadIdx.x;  // T*B*64
  if (i >= Tc * Bc * LHc) return;
  int d  = i & 63;
  int bt = i >> 6;          // t*16 + b
  int b  = bt & 15;
  int t  = bt >> 4;
  xs_f[i] = fused[((size_t)b * Tc + t) * LHc + d];
  int rt = clampi_(lengths[b] - 1 - t, 0, Tc - 1);
  xs_b[i] = fused[((size_t)b * Tc + rt) * LHc + d];
}

__global__ void bias_sum_kernel(const float* a, const float* b, float* o, int n)
{
  int i = blockIdx.x * blockDim.x + threadIdx.x;
  if (i < n) o[i] = a[i] + b[i];
}

// ============================================================
// Kernel 5: sequential LSTM (2 blocks: dir 0 = fwd, 1 = bwd).
//           gates = gin[t] (precomputed x@Wih^T+b) + h@Whh^T via WMMA f32.
// ============================================================
__global__ __launch_bounds__(512) void lstm_seq_kernel(
    const float* __restrict__ gin_f, const float* __restrict__ gin_b,
    const float* __restrict__ whh_f, const float* __restrict__ whh_b,
    const int* __restrict__ lengths, float* __restrict__ last)
{
  __shared__ float sH[Bc * LHc];
  __shared__ float sC[Bc * LHc];
  __shared__ float sG[Bc * NG];

  const int dir = blockIdx.x;
  const float* gin = dir ? gin_b : gin_f;
  const float* whh = dir ? whh_b : whh_f;   // [256][64]
  const int tid  = threadIdx.x;
  const int lane = tid & 31;
  const int wv   = __builtin_amdgcn_readfirstlane(tid >> 5);  // 0..15
  const int hi   = (lane >> 4) & 1;
  const int ln   = lane & 15;
  const int g0   = wv * 16;                 // scalar gate-tile base

  for (int i = tid; i < Bc * LHc; i += 512) { sH[i] = 0.0f; sC[i] = 0.0f; }
  __syncthreads();

  for (int t = 0; t < Tc; ++t) {
    // init accumulator with precomputed input projection (+ biases)
    v8f acc;
#pragma unroll
    for (int v = 0; v < 8; ++v) {
      int m = v + hi * 8;
      acc[v] = gin[((size_t)t * Bc + m) * NG + g0 + ln];
    }
    // h[16x64] @ whh^T tile -> 16 K-chunks of WMMA f32 16x16x4
    for (int kc = 0; kc < LHc; kc += 4) {
      v2f a, b;
#pragma unroll
      for (int v = 0; v < 2; ++v) {
        int k = kc + hi * 2 + v;
        a[v] = sH[ln * LHc + k];                       // A(batch=ln, k)
        b[v] = whh[(size_t)(g0 + ln) * LHc + k];       // B(k, gate=ln)
      }
      acc = __builtin_amdgcn_wmma_f32_16x16x4_f32(false, a, false, b, (short)0, acc, false, false);
    }
#pragma unroll
    for (int v = 0; v < 8; ++v) {
      int m = v + hi * 8;
      sG[m * NG + g0 + ln] = acc[v];
    }
    __syncthreads();

    // cell update, gate order i,f,g,o
    for (int idx = tid; idx < Bc * LHc; idx += 512) {
      int bb = idx >> 6, d = idx & 63;
      float gi = sG[bb * NG + d];
      float gf = sG[bb * NG + LHc + d];
      float gg = sG[bb * NG + 2 * LHc + d];
      float go = sG[bb * NG + 3 * LHc + d];
      float c = sigmoidf_(gf) * sC[idx] + sigmoidf_(gi) * tanhf(gg);
      float h = sigmoidf_(go) * tanhf(c);
      sC[idx] = c;
      sH[idx] = h;
      if (dir == 0) { if (t == lengths[bb] - 1) last[bb * 128 + d] = h; }
      else          { if (t == 0)               last[bb * 128 + LHc + d] = h; }
    }
    __syncthreads();
  }
}

// ============================================================
// Kernel 6: final FC  out[16][10] = last[16][128] @ fc_w^T + fc_b
// ============================================================
__global__ void fc_kernel(const float* __restrict__ last,
                          const float* __restrict__ fcw,
                          const float* __restrict__ fcb,
                          float* __restrict__ out)
{
  int i = blockIdx.x * blockDim.x + threadIdx.x;
  if (i >= Bc * NCc) return;
  int b = i / NCc, n = i % NCc;
  float acc = fcb[n];
  for (int k = 0; k < 2 * LHc; ++k) acc += last[b * 128 + k] * fcw[n * 128 + k];
  out[i] = acc;
}

// ============================================================
extern "C" void kernel_launch(void* const* d_in, const int* in_sizes, int n_in,
                              void* d_out, int out_size, void* d_ws, size_t ws_size,
                              hipStream_t stream)
{
  const float* x_real   = (const float*)d_in[0];
  const float* x_imag   = (const float*)d_in[1];
  const int*   lengths  = (const int*)  d_in[2];
  const float* c1r_w = (const float*)d_in[3];
  const float* c1r_b = (const float*)d_in[4];
  const float* c1i_w = (const float*)d_in[5];
  const float* c1i_b = (const float*)d_in[6];
  const float* c2r_w = (const float*)d_in[7];
  const float* c2r_b = (const float*)d_in[8];
  const float* c2i_w = (const float*)d_in[9];
  const float* c2i_b = (const float*)d_in[10];
  const float* fus_w = (const float*)d_in[11];
  const float* fus_b = (const float*)d_in[12];
  const float* wf_ih = (const float*)d_in[13];
  const float* wf_hh = (const float*)d_in[14];
  const float* bf_ih = (const float*)d_in[15];
  const float* bf_hh = (const float*)d_in[16];
  const float* wb_ih = (const float*)d_in[17];
  const float* wb_hh = (const float*)d_in[18];
  const float* bb_ih = (const float*)d_in[19];
  const float* bb_hh = (const float*)d_in[20];
  const float* fc_w  = (const float*)d_in[21];
  const float* fc_b  = (const float*)d_in[22];
  float* out = (float*)d_out;

  // workspace carve-up (floats)
  float* ws = (float*)d_ws;
  size_t o = 0;
  float* pooled_r = ws + o; o += (size_t)BT * C2c;
  float* pooled_m = ws + o; o += (size_t)BT * C2c;
  float* feat     = ws + o; o += (size_t)BT * 4 * C2c;
  float* fused    = ws + o; o += (size_t)BT * LHc;      // [b][t][64]
  float* xs_f     = ws + o; o += (size_t)Tc * Bc * LHc; // [t][b][64]
  float* xs_b     = ws + o; o += (size_t)Tc * Bc * LHc;
  float* gin_f    = ws + o; o += (size_t)Tc * Bc * NG;
  float* gin_b    = ws + o; o += (size_t)Tc * Bc * NG;
  float* lastbuf  = ws + o; o += (size_t)Bc * 2 * LHc;
  float* bsum_f   = ws + o; o += NG;
  float* bsum_b   = ws + o; o += NG;
  (void)o; (void)ws_size; (void)n_in; (void)in_sizes; (void)out_size;

  // conv stacks (real + imag), fused conv1->conv2->pool
  conv_pool_kernel<<<BT, 256, 0, stream>>>(x_real, c1r_w, c1r_b, c2r_w, c2r_b, pooled_r);
  conv_pool_kernel<<<BT, 256, 0, stream>>>(x_imag, c1i_w, c1i_b, c2i_w, c2i_b, pooled_m);

  // features
  feat_kernel<<<(BT * C2c + 255) / 256, 256, 0, stream>>>(pooled_r, pooled_m, feat);

  // fused = relu(feat @ fus_w^T + fus_b): M=1536, N=64, K=128 -> 384 tiles
  gemm_wmma_f32<<<(384 + 3) / 4, 128, 0, stream>>>(feat, fus_w, fus_b, fused, BT, LHc, 4 * C2c, 1);

  // combined LSTM biases
  bias_sum_kernel<<<1, NG, 0, stream>>>(bf_ih, bf_hh, bsum_f, NG);
  bias_sum_kernel<<<1, NG, 0, stream>>>(bb_ih, bb_hh, bsum_b, NG);

  // t-major + length-reversed sequences
  seq_prep_kernel<<<(Tc * Bc * LHc + 255) / 256, 256, 0, stream>>>(fused, lengths, xs_f, xs_b);

  // input projections for both directions: M=1536, N=256, K=64 -> 1536 tiles
  gemm_wmma_f32<<<(1536 + 3) / 4, 128, 0, stream>>>(xs_f, wf_ih, bsum_f, gin_f, Tc * Bc, NG, LHc, 0);
  gemm_wmma_f32<<<(1536 + 3) / 4, 128, 0, stream>>>(xs_b, wb_ih, bsum_b, gin_b, Tc * Bc, NG, LHc, 0);

  // sequential recurrences (fwd + bwd in parallel blocks)
  lstm_seq_kernel<<<2, 512, 0, stream>>>(gin_f, gin_b, wf_hh, wb_hh, lengths, lastbuf);

  // classifier
  fc_kernel<<<1, Bc * NCc, 0, stream>>>(lastbuf, fc_w, fc_b, out);
}
